// Capsule_55688545960037
// MI455X (gfx1250) — compile-verified, matrix operand order checked by reading, not run on hardware
//
#include <hip/hip_runtime.h>
#include <hip/hip_bf16.h>

// ---------------- problem constants ----------------
#define B_   32
#define IC_  2048
#define IV_  16
#define OC_  32
#define OV_  32
#define KI_  (IC_ * IV_)   // 32768 flattened contraction (k,i)

typedef __attribute__((ext_vector_type(16))) __bf16       v16bf;
typedef __attribute__((ext_vector_type(8)))  float        v8f;
typedef __attribute__((ext_vector_type(4)))  unsigned int v4u;
typedef __attribute__((ext_vector_type(2)))  unsigned int v2u;

union Frag { v16bf v; unsigned int u[8]; v4u q[2]; };

// f32 -> bf16 round-to-nearest-even (bit trick)
static __device__ inline unsigned int f2bf(float f) {
  unsigned int u = __float_as_uint(f);
  u = u + 0x7FFFu + ((u >> 16) & 1u);
  return u >> 16;
}

// ---------------- fills ----------------
__global__ void k_fill(float* __restrict__ p, int n, float val) {
  int i = blockIdx.x * blockDim.x + threadIdx.x;
  for (; i < n; i += gridDim.x * blockDim.x) p[i] = val;
}

// ---------------- s-pass ----------------
// s[b,o,j] += sum_{kf in segment} (c[b,o,kf/16] * x[b,kf]) * W[o,kf,j]
// grid: (OC_, SP_KSEG), block: 128 (4 waves, one 16x16 output tile each)
#define SP_KSEG  16
#define SP_SEG   (KI_ / SP_KSEG)   // 2048
#define SP_CHUNK 256

__global__ void __launch_bounds__(128)
k_s_pass(const float* __restrict__ x, const float* __restrict__ W,
         const float* __restrict__ c, float* __restrict__ s)
{
  // A: bf16 x rows, K-contiguous; row stride 264 shorts = 528B (16B-aligned)
  __shared__ unsigned short Alds[B_][SP_CHUNK + 8];
  // B: pair-packed transposed W: BpT[j][K2] = {bf16(K+1)<<16 | bf16(K)},
  // row stride 132 uints = 528B (16B-aligned)
  __shared__ unsigned int BpT[OV_][SP_CHUNK / 2 + 4];

  const int o    = blockIdx.x;
  const int kf0  = blockIdx.y * SP_SEG;
  const int tid  = threadIdx.x;
  const int wv   = tid >> 5, lane = tid & 31;
  const int m0   = (wv & 1) * 16, n0 = (wv >> 1) * 16;
  const int mrow = lane & 15, h = lane >> 4;

  v8f acc = {0.f, 0.f, 0.f, 0.f, 0.f, 0.f, 0.f, 0.f};

  for (int ch = 0; ch < SP_SEG / SP_CHUNK; ++ch) {
    const int kfc = kf0 + ch * SP_CHUNK;
    // A fill: float4 x loads, one c per quad (quad never straddles a capsule)
    for (int idx = tid; idx < B_ * (SP_CHUNK / 4); idx += 128) {
      int b   = idx >> 6;                       // / (SP_CHUNK/4)
      int kl4 = (idx & (SP_CHUNK / 4 - 1)) * 4;
      int kf  = kfc + kl4;
      float4 xv = *(const float4*)&x[b * KI_ + kf];
      float  cs = c[(b * OC_ + o) * IC_ + (kf >> 4)];
      v2u pk = { f2bf(xv.x * cs) | (f2bf(xv.y * cs) << 16),
                 f2bf(xv.z * cs) | (f2bf(xv.w * cs) << 16) };
      *(v2u*)&Alds[b][kl4] = pk;                // 8B packed store
    }
    // B fill: two float4 W rows -> 4 pair-packed uints
    for (int idx = tid; idx < (SP_CHUNK / 2) * (OV_ / 4); idx += 128) {
      int K2 = idx >> 3;
      int j4 = (idx & 7) * 4;
      const float* wp = &W[((size_t)o * KI_ + (size_t)(kfc + 2 * K2)) * OV_ + j4];
      float4 w0 = *(const float4*)wp;
      float4 w1 = *(const float4*)(wp + OV_);
      BpT[j4 + 0][K2] = f2bf(w0.x) | (f2bf(w1.x) << 16);
      BpT[j4 + 1][K2] = f2bf(w0.y) | (f2bf(w1.y) << 16);
      BpT[j4 + 2][K2] = f2bf(w0.z) | (f2bf(w1.z) << 16);
      BpT[j4 + 3][K2] = f2bf(w0.w) | (f2bf(w1.w) << 16);
    }
    __syncthreads();
    for (int ks = 0; ks < SP_CHUNK / 32; ++ks) {
      Frag af, bf;
      af.q[0] = *(const v4u*)&Alds[m0 + mrow][ks * 32 + 8 * h];
      af.q[1] = *(const v4u*)&Alds[m0 + mrow][ks * 32 + 16 + 8 * h];
      bf.q[0] = *(const v4u*)&BpT[n0 + mrow][ks * 16 + 4 * h];
      bf.q[1] = *(const v4u*)&BpT[n0 + mrow][ks * 16 + 8 + 4 * h];
      acc = __builtin_amdgcn_wmma_f32_16x16x32_bf16(false, af.v, false, bf.v,
                                                    (short)0, acc, false, false);
    }
    __syncthreads();
  }
#pragma unroll
  for (int r = 0; r < 8; ++r) {      // C layout: m = r + 8h, n = lane%16
    int b = m0 + r + 8 * h;
    int j = n0 + mrow;
    atomicAdd(&s[(b * OC_ + o) * OV_ + j], acc[r]);
  }
}

// ---------------- squash ----------------
__global__ void k_squash(const float* __restrict__ s, float* __restrict__ dst) {
  int vec = blockIdx.x;
  int j   = threadIdx.x;
  float v  = s[vec * OV_ + j];
  float n2 = v * v;
#pragma unroll
  for (int m = 16; m >= 1; m >>= 1) n2 += __shfl_xor(n2, m, 32);
  float n = sqrtf(n2);
  dst[vec * OV_ + j] = v * (n / (1.0f + n2));
}

// ---------------- agreement update ----------------
// blog[b,o,k] += sum_i x[b,k,i] * (W[o,k,i,:] . a[b,o,:])
// WA[kf,b] = W[o] . a^T via WMMA (K=j=32 exactly), then VALU dot with x.
// grid: (OC_, IC_/BU_CAPS), block 128 (4 waves)
#define BU_CAPS  64
#define BU_STAGE 8     // capsules per LDS stage (128 kf rows)

__global__ void __launch_bounds__(128)
k_bupdate(const float* __restrict__ x, const float* __restrict__ W,
          const float* __restrict__ a, float* __restrict__ blog)
{
  // A: bf16 W rows [kf][j], row stride 40 shorts = 80B (16B-aligned)
  __shared__ unsigned short Wlds[BU_STAGE * 16][OV_ + 8];
  __shared__ float          xT[BU_STAGE * 16][B_ + 1];
  // B: pair-packed a^T: apT[b][j2], row stride 20 uints = 80B (16B-aligned)
  __shared__ unsigned int   apT[B_][OV_ / 2 + 4];

  const int o     = blockIdx.x;
  const int kcap0 = blockIdx.y * BU_CAPS;
  const int tid   = threadIdx.x;
  const int wv    = tid >> 5, lane = tid & 31;
  const int mrow  = lane & 15, h = lane >> 4;

  for (int idx = tid; idx < B_ * (OV_ / 4); idx += 128) {   // a -> apT
    int b = idx >> 3, j4 = (idx & 7) * 4;
    float4 av = *(const float4*)&a[(b * OC_ + o) * OV_ + j4];
    v2u pk = { f2bf(av.x) | (f2bf(av.y) << 16),
               f2bf(av.z) | (f2bf(av.w) << 16) };
    *(v2u*)&apT[b][j4 >> 1] = pk;
  }

  for (int st = 0; st < BU_CAPS / BU_STAGE; ++st) {
    const int cap0 = kcap0 + st * BU_STAGE;
    const int kfc  = cap0 * IV_;
    __syncthreads();
    for (int idx = tid; idx < BU_STAGE * 16 * (OV_ / 4); idx += 128) {  // W
      int kl = idx >> 3, j4 = (idx & 7) * 4;
      float4 w0 = *(const float4*)&W[((size_t)o * KI_ + (size_t)(kfc + kl)) * OV_ + j4];
      v2u pk = { f2bf(w0.x) | (f2bf(w0.y) << 16),
                 f2bf(w0.z) | (f2bf(w0.w) << 16) };
      *(v2u*)&Wlds[kl][j4] = pk;
    }
    for (int idx = tid; idx < B_ * (BU_STAGE * 16 / 4); idx += 128) {   // x^T
      int b = idx >> 5, kl4 = (idx & 31) * 4;
      float4 xv = *(const float4*)&x[b * KI_ + kfc + kl4];
      xT[kl4 + 0][b] = xv.x;
      xT[kl4 + 1][b] = xv.y;
      xT[kl4 + 2][b] = xv.z;
      xT[kl4 + 3][b] = xv.w;
    }
    __syncthreads();
    for (int cc = 0; cc < 2; ++cc) {
      int capL = wv * 2 + cc;                 // capsule within stage
      for (int n0 = 0; n0 < 32; n0 += 16) {   // b-halves
        Frag af, bf;
        af.q[0] = *(const v4u*)&Wlds[capL * 16 + mrow][8 * h];
        af.q[1] = *(const v4u*)&Wlds[capL * 16 + mrow][16 + 8 * h];
        bf.q[0] = *(const v4u*)&apT[n0 + mrow][4 * h];
        bf.q[1] = *(const v4u*)&apT[n0 + mrow][8 + 4 * h];
        v8f wa = {0.f, 0.f, 0.f, 0.f, 0.f, 0.f, 0.f, 0.f};
        wa = __builtin_amdgcn_wmma_f32_16x16x32_bf16(false, af.v, false, bf.v,
                                                     (short)0, wa, false, false);
        float partial = 0.f;
#pragma unroll
        for (int r = 0; r < 8; ++r)
          partial += xT[capL * 16 + r + 8 * h][n0 + mrow] * wa[r];
        partial += __shfl_xor(partial, 16, 32);   // join rows 0-7 with 8-15
        if (h == 0) {
          int b = n0 + mrow;
          int k = cap0 + capL;
          blog[(b * OC_ + o) * IC_ + k] += partial;   // unique writer
        }
      }
    }
  }
}

// ---------------- softmax over out-capsules ----------------
__global__ void k_softmax(const float* __restrict__ blog, float* __restrict__ c) {
  int idx = blockIdx.x * blockDim.x + threadIdx.x;   // over B_*IC_
  if (idx >= B_ * IC_) return;
  int b = idx / IC_, k = idx - b * IC_;
  const float* p = blog + (size_t)b * OC_ * IC_ + k;
  float mx = -1e30f;
#pragma unroll
  for (int o = 0; o < OC_; ++o) mx = fmaxf(mx, p[o * IC_]);
  float e[OC_], sum = 0.f;
#pragma unroll
  for (int o = 0; o < OC_; ++o) { e[o] = __expf(p[o * IC_] - mx); sum += e[o]; }
  float inv = 1.0f / sum;
  float* q = c + (size_t)b * OC_ * IC_ + k;
#pragma unroll
  for (int o = 0; o < OC_; ++o) q[o * IC_] = e[o] * inv;
}

// ---------------- host orchestration ----------------
extern "C" void kernel_launch(void* const* d_in, const int* in_sizes, int n_in,
                              void* d_out, int out_size, void* d_ws, size_t ws_size,
                              hipStream_t stream) {
  (void)in_sizes; (void)n_in; (void)out_size; (void)ws_size;
  const float* x = (const float*)d_in[0];   // [B, IC, IV]
  const float* W = (const float*)d_in[1];   // [OC, IC, IV, OV]
  float* out  = (float*)d_out;              // [B, OC, OV]

  float* blog = (float*)d_ws;                       // B*OC*IC
  float* c    = blog + (size_t)B_ * OC_ * IC_;      // B*OC*IC
  float* s    = c    + (size_t)B_ * OC_ * IC_;      // B*OC*OV
  float* a    = s    + (size_t)B_ * OC_ * OV_;      // B*OC*OV

  const dim3 gs(OC_, SP_KSEG);
  const dim3 gb(OC_, IC_ / BU_CAPS);
  const int NS = B_ * OC_ * OV_;
  const int NB = B_ * OC_ * IC_;

  k_fill<<<256, 256, 0, stream>>>(blog, NB, 0.0f);
  k_fill<<<256, 256, 0, stream>>>(c, NB, 1.0f / 32.0f);  // softmax(0) exactly
  k_fill<<<64, 256, 0, stream>>>(s, NS, 0.0f);

  // iter 0 (uniform c)
  k_s_pass<<<gs, 128, 0, stream>>>(x, W, c, s);
  k_squash<<<B_ * OC_, 32, 0, stream>>>(s, a);
  k_bupdate<<<gb, 128, 0, stream>>>(x, W, a, blog);
  k_softmax<<<(B_ * IC_ + 255) / 256, 256, 0, stream>>>(blog, c);

  // iter 1
  k_fill<<<64, 256, 0, stream>>>(s, NS, 0.0f);
  k_s_pass<<<gs, 128, 0, stream>>>(x, W, c, s);
  k_squash<<<B_ * OC_, 32, 0, stream>>>(s, a);
  k_bupdate<<<gb, 128, 0, stream>>>(x, W, a, blog);
  k_softmax<<<(B_ * IC_ + 255) / 256, 256, 0, stream>>>(blog, c);

  // iter 2 (final): v = squash(s)
  k_fill<<<64, 256, 0, stream>>>(s, NS, 0.0f);
  k_s_pass<<<gs, 128, 0, stream>>>(x, W, c, s);
  k_squash<<<B_ * OC_, 32, 0, stream>>>(s, out);
}
